// skip_LSTMCell_18915035971747
// MI455X (gfx1250) — compile-verified, matrix operand order checked by reading, not run on hardware
//
#include <hip/hip_runtime.h>

// ---------------------------------------------------------------------------
// skip-LSTM cell, fused for MI455X (gfx1250, wave32, WMMA bf16 -> f32 accum)
// ---------------------------------------------------------------------------
#define BATCH 2048
#define UNITS 1024
#define DIN   1024
#define SKIPN 24

#define BM 128
#define BN 128
#define BK 32
#define LDT 40          // padded LDS row (bf16 elements); even, col%4==0 -> 8B aligned

typedef __attribute__((ext_vector_type(16))) __bf16 v16bf;
typedef __attribute__((ext_vector_type(8)))  float  v8f;
typedef __attribute__((ext_vector_type(4)))  float  f32x4;
typedef __attribute__((ext_vector_type(4)))  __bf16 bf16x4;

// One uniform K-phase of the tiled GEMM: constant strides, pointer-bumped.
// A source: [BM x ksteps*BK] slab starting at aptr (row stride lda)
// B source: [ksteps*BK x BN] slab starting at bptr (row stride ldb)
__device__ __forceinline__
void gemm_phase(const float* __restrict__ aptr, int lda,
                const float* __restrict__ bptr, int ldb,
                int ksteps, int t,
                __bf16* __restrict__ As, __bf16* __restrict__ Bs,
                const __bf16* __restrict__ afrag_base,   // &As[(wm+ln)*LDT]
                const __bf16* __restrict__ bfrag_base,   // &Bs[(wn+ln)*LDT]
                int klo, v8f acc[4][2])
{
    // per-thread staging offsets (loop-invariant)
    long aoff[4], boff[4];
    int  acol[4], bcol[4], bkrow[4];
    #pragma unroll
    for (int i = 0; i < 4; ++i) {
        int idx = t + i * 256;                 // 0..1023
        int arow = idx >> 3;                   // /8
        acol[i]  = (idx & 7) << 2;             // *4
        aoff[i]  = (long)arow * lda + acol[i];
        bkrow[i] = idx >> 5;                   // /32 -> 0..31
        bcol[i]  = (idx & 31) << 2;            // 0..124
        boff[i]  = (long)bkrow[i] * ldb + bcol[i];
        (void)acol[i];
    }
    __bf16* asdst[4];
    #pragma unroll
    for (int i = 0; i < 4; ++i) {
        int idx = t + i * 256;
        asdst[i] = &As[(idx >> 3) * LDT + ((idx & 7) << 2)];
    }

    for (int ks = 0; ks < ksteps; ++ks) {
        // ---- stage A tile (BM x BK f32 -> bf16)
        #pragma unroll
        for (int i = 0; i < 4; ++i) {
            f32x4 v = *(const f32x4*)(aptr + aoff[i]);
            *(bf16x4*)asdst[i] = __builtin_convertvector(v, bf16x4);
        }
        // ---- stage B tile (BK x BN f32 -> bf16, transposed into Bs[n][k])
        #pragma unroll
        for (int i = 0; i < 4; ++i) {
            f32x4 v = *(const f32x4*)(bptr + boff[i]);
            bf16x4 b = __builtin_convertvector(v, bf16x4);
            Bs[(bcol[i] + 0) * LDT + bkrow[i]] = b.x;
            Bs[(bcol[i] + 1) * LDT + bkrow[i]] = b.y;
            Bs[(bcol[i] + 2) * LDT + bkrow[i]] = b.z;
            Bs[(bcol[i] + 3) * LDT + bkrow[i]] = b.w;
        }
        __syncthreads();

        #pragma unroll
        for (int ni = 0; ni < 2; ++ni) {
            union { v16bf v; unsigned u[8]; } bf;
            const unsigned short* brow =
                (const unsigned short*)(bfrag_base + ni * 16 * LDT);
            #pragma unroll
            for (int v = 0; v < 4; ++v) {
                bf.u[v]     = *(const unsigned*)(brow + klo + 2 * v);
                bf.u[v + 4] = *(const unsigned*)(brow + 16 + klo + 2 * v);
            }
            #pragma unroll
            for (int mi = 0; mi < 4; ++mi) {
                union { v16bf v; unsigned u[8]; } af;
                const unsigned short* arow =
                    (const unsigned short*)(afrag_base + mi * 16 * LDT);
                #pragma unroll
                for (int v = 0; v < 4; ++v) {
                    af.u[v]     = *(const unsigned*)(arow + klo + 2 * v);
                    af.u[v + 4] = *(const unsigned*)(arow + 16 + klo + 2 * v);
                }
                acc[mi][ni] = __builtin_amdgcn_wmma_f32_16x16x32_bf16(
                    false, af.v, false, bf.v, (short)0, acc[mi][ni], false, false);
            }
        }
        __syncthreads();

        aptr += BK;                 // constant bump, K advances along A rows
        bptr += (long)BK * ldb;     // constant bump, K advances along B rows
    }
}

// Fused GEMM over virtual output [BATCH x 5*UNITS]:
//   cols [0,4096): gates pre-act = inputs@kernel + h_part_g@rkernel + bias  -> out_g
//   cols [4096,5120): skip pre-act = prev_h[:, :U]@kernel2 + bias2          -> out_h
__global__ __launch_bounds__(256)
void fused_gates_gemm(const float* __restrict__ inputs,
                      const float* __restrict__ h_tm1,
                      const float* __restrict__ prev_h,
                      const float* __restrict__ wk,     // kernel   [D,4U]
                      const float* __restrict__ wr,     // rkernel  [U,4U]
                      const float* __restrict__ w2,     // kernel2  [U,U]
                      const float* __restrict__ bias,   // [4U]
                      const float* __restrict__ bias2,  // [U]
                      float* __restrict__ out_h,        // [B,U]   (skip pre-act)
                      float* __restrict__ out_g)        // [B,4U]  (gate pre-act)
{
    __shared__ __bf16 As[BM * LDT];   // [m][k]
    __shared__ __bf16 Bs[BN * LDT];   // [n][k]  (B transposed)

    const int t       = threadIdx.x;
    const int rowbase = blockIdx.y * BM;      // 0..1920
    const int colbase = blockIdx.x * BN;      // 0..4992

    const bool is_skip = (colbase >= 4 * UNITS);
    const int  outcol  = is_skip ? (colbase - 4 * UNITS) : colbase;

    // wave32 layout: 8 waves, 2(M) x 4(N); each wave 64x32 -> 4x2 WMMA tiles
    const int w    = t >> 5;
    const int lane = t & 31;
    const int wm   = (w >> 2) * 64;
    const int wn   = (w & 3) * 32;
    const int ln   = lane & 15;
    const int hi   = lane >> 4;       // 0 | 1
    const int klo  = hi * 8;          // 16-bit A/B fragment K-base per half-wave

    const __bf16* afrag_base = &As[(wm + ln) * LDT];
    const __bf16* bfrag_base = &Bs[(wn + ln) * LDT];

    v8f acc[4][2];
    #pragma unroll
    for (int i = 0; i < 4; ++i)
        #pragma unroll
        for (int j = 0; j < 2; ++j)
            acc[i][j] = (v8f)0.f;

    if (is_skip) {
        // skip: prev_h[:, :U] @ kernel2, K = UNITS
        gemm_phase(prev_h + (long)rowbase * (UNITS * SKIPN), UNITS * SKIPN,
                   w2 + outcol, UNITS,
                   UNITS / BK, t, As, Bs, afrag_base, bfrag_base, klo, acc);
    } else {
        const int g = colbase >> 10;          // gate index 0..3
        // phase 1: inputs @ kernel, K = DIN
        gemm_phase(inputs + (long)rowbase * DIN, DIN,
                   wk + colbase, 4 * UNITS,
                   DIN / BK, t, As, Bs, afrag_base, bfrag_base, klo, acc);
        // phase 2: h_part_g @ rkernel, K = UNITS
        gemm_phase(h_tm1 + (long)rowbase * (4 * UNITS) + (g << 10), 4 * UNITS,
                   wr + colbase, 4 * UNITS,
                   UNITS / BK, t, As, Bs, afrag_base, bfrag_base, klo, acc);
    }

    // ---- epilogue: +bias, store pre-activations
    float*       outp = is_skip ? out_h : out_g;
    const int    ldo  = is_skip ? UNITS : 4 * UNITS;
    const float* bp   = is_skip ? bias2 : bias;

    #pragma unroll
    for (int ni = 0; ni < 2; ++ni) {
        int n = outcol + wn + ni * 16 + ln;
        float bv = bp[n];
        #pragma unroll
        for (int mi = 0; mi < 4; ++mi) {
            int mrow = rowbase + wm + mi * 16 + hi * 8;
            #pragma unroll
            for (int r = 0; r < 8; ++r)
                outp[(long)(mrow + r) * ldo + n] = acc[mi][ni][r] + bv;
        }
    }
}

__device__ __forceinline__ float sigm(float x) { return 1.f / (1.f + __expf(-x)); }

// Elementwise LSTM epilogue: reads pre-activations in-place from d_out regions.
__global__ __launch_bounds__(256)
void lstm_epilogue(const float* __restrict__ c_tm1,
                   const float* __restrict__ step,
                   const float* __restrict__ s0p,
                   float* __restrict__ out_h,     // in: skip pre-act, out: h
                   float* __restrict__ out_g,     // in: gate pre-act, out: [i,f,c,o]
                   float* __restrict__ out_c,
                   float* __restrict__ out_step,
                   float* __restrict__ out_prev)
{
    int idx = blockIdx.x * 256 + threadIdx.x;     // B*U threads
    if (idx >= BATCH * UNITS) return;
    int b = idx >> 10;
    int u = idx & (UNITS - 1);
    float s0 = s0p[0];
    long gb = (long)b * (4 * UNITS);

    float pi = out_g[gb + u];
    float pf = out_g[gb + UNITS + u];
    float pc = out_g[gb + 2 * UNITS + u];
    float po = out_g[gb + 3 * UNITS + u];

    float i = sigm(pi);
    float f = sigm(pf);
    float c = f * c_tm1[idx] + i * tanhf(pc);
    float o = sigm(po);
    float hh = o * tanhf(c);
    float sk = sigm(out_h[idx]);
    float h  = s0 * hh + (1.f - s0) * sk;

    out_h[idx] = h;
    out_g[gb + u]             = i;
    out_g[gb + UNITS + u]     = f;
    out_g[gb + 2 * UNITS + u] = c;
    out_g[gb + 3 * UNITS + u] = o;
    out_c[idx] = c;
    out_prev[(long)b * (UNITS * SKIPN) + (SKIPN - 1) * UNITS + u] = h;
    if (u == 0) out_step[b] = step[b] + 1.f;
}

// Shift copy: new_prev_h[:, :(SKIP-1)*U] = prev_h[:, U:]
__global__ __launch_bounds__(256)
void shift_prev(const float* __restrict__ prev_h, float* __restrict__ out_prev)
{
    const long total = (long)BATCH * (SKIPN - 1) * UNITS / 4;   // float4 count
    long idx = (long)blockIdx.x * 256 + threadIdx.x;
    if (idx >= total) return;
    long e = idx * 4;
    int b = (int)(e / ((SKIPN - 1) * UNITS));
    int r = (int)(e % ((SKIPN - 1) * UNITS));
    *(float4*)(out_prev + (long)b * (UNITS * SKIPN) + r) =
        *(const float4*)(prev_h + (long)b * (UNITS * SKIPN) + UNITS + r);
}

extern "C" void kernel_launch(void* const* d_in, const int* in_sizes, int n_in,
                              void* d_out, int out_size, void* d_ws, size_t ws_size,
                              hipStream_t stream)
{
    const float* inputs  = (const float*)d_in[0];
    const float* h_tm1   = (const float*)d_in[1];
    const float* c_tm1   = (const float*)d_in[2];
    const float* step    = (const float*)d_in[3];
    const float* prev_h  = (const float*)d_in[4];
    const float* wk      = (const float*)d_in[5];
    const float* wr      = (const float*)d_in[6];
    const float* w2      = (const float*)d_in[7];
    const float* bias    = (const float*)d_in[8];
    const float* bias2   = (const float*)d_in[9];
    const float* s0      = (const float*)d_in[10];

    float* out      = (float*)d_out;
    float* out_h    = out;                                   // [B,U]
    float* out_g    = out + (long)BATCH * UNITS;             // [B,4U]
    float* out_c    = out + (long)BATCH * UNITS * 5;         // [B,U]
    float* out_step = out + (long)BATCH * UNITS * 6;         // [B,1]
    float* out_prev = out_step + BATCH;                      // [B,U*SKIP]

    dim3 grid(5 * UNITS / BN, BATCH / BM);                   // 40 x 16
    fused_gates_gemm<<<grid, 256, 0, stream>>>(inputs, h_tm1, prev_h, wk, wr, w2,
                                               bias, bias2, out_h, out_g);

    lstm_epilogue<<<(BATCH * UNITS) / 256, 256, 0, stream>>>(c_tm1, step, s0,
                                                             out_h, out_g, out_c,
                                                             out_step, out_prev);

    long cp = (long)BATCH * (SKIPN - 1) * UNITS / 4;
    shift_prev<<<(unsigned)((cp + 255) / 256), 256, 0, stream>>>(prev_h, out_prev);
}
